// GraphDistillationLoss_68917045231788
// MI455X (gfx1250) — matrix operand order repaired and checked by reading, not compiled, not run on hardware
//
#include <hip/hip_runtime.h>
#include <stdint.h>

#define N_NODES 8192
#define D_FEAT  128
#define K_SPLIT 8
#define EPS_F   1e-6f

typedef __attribute__((ext_vector_type(16))) __bf16 v16bf;
typedef __attribute__((ext_vector_type(8)))  float  v8f;
typedef __attribute__((ext_vector_type(2)))  float  v2f;

__device__ __forceinline__ unsigned short f2bf_bits(float x) {
    unsigned u = __builtin_bit_cast(unsigned, x);
    unsigned r = (u + 0x7FFFu + ((u >> 16) & 1u)) >> 16;   // RNE
    return (unsigned short)r;
}
__device__ __forceinline__ float bf2f(unsigned short b) {
    return __builtin_bit_cast(float, ((unsigned)b) << 16);
}

union V16U { uint4 q[2]; v16bf v; unsigned short s[16]; unsigned u[8]; };

__device__ __forceinline__ v16bf load_v16(const unsigned short* p) {
    V16U u;
    u.q[0] = *(const uint4*)p;
    u.q[1] = *(const uint4*)(p + 8);
    return u.v;
}

__device__ __forceinline__ v8f wmma_bf16(v16bf a, v16bf b, v8f c) {
    return __builtin_amdgcn_wmma_f32_16x16x32_bf16(false, a, false, b, (short)0, c, false, false);
}

// ---------------------------------------------------------------------------
// Kernel 1: d = s - t; bf16 hi/lo transposed [f][k]; per-row sq; sum(sq);
// also fills the bf16-ones buffer used as the row-count WMMA column.
// ---------------------------------------------------------------------------
__global__ __launch_bounds__(128) void prep_kernel(
    const float* __restrict__ s, const float* __restrict__ t,
    float* __restrict__ sqArr,
    unsigned short* __restrict__ dHiT, unsigned short* __restrict__ dLoT,
    unsigned short* __restrict__ sqHi, unsigned short* __restrict__ sqLo,
    unsigned short* __restrict__ onesBuf, float* __restrict__ acc)
{
    const int i = blockIdx.x;
    const int f = threadIdx.x;
    const size_t idx = (size_t)i * D_FEAT + f;
    float d = s[idx] - t[idx];
    unsigned short hb = f2bf_bits(d);
    unsigned short lb = f2bf_bits(d - bf2f(hb));
    dHiT[(size_t)f * N_NODES + i] = hb;   // scatter; 4 MB total, L2 absorbs it
    dLoT[(size_t)f * N_NODES + i] = lb;

    float e = d * d;
    #pragma unroll
    for (int m = 16; m > 0; m >>= 1) e += __shfl_xor(e, m, 32);
    __shared__ float red[4];
    const int lane = threadIdx.x & 31;
    const int wv   = threadIdx.x >> 5;
    if (lane == 0) red[wv] = e;
    __syncthreads();
    if (threadIdx.x == 0) {
        float tot = red[0] + red[1] + red[2] + red[3];
        sqArr[i] = tot;
        unsigned short sh = f2bf_bits(tot);
        sqHi[i] = sh;
        sqLo[i] = f2bf_bits(tot - bf2f(sh));
        onesBuf[i] = (unsigned short)0x3F80;   // bf16 1.0
        atomicAdd(&acc[2], tot);
    }
}

// ---------------------------------------------------------------------------
// Kernel 2: stream adj once; per wave: 16-row strip x 1024-K slice.
// C[16x128] += mask(16x32) @ dT(32x128) via bf16 WMMA (hi+lo), plus one
// extra WMMA tile: col0=sq_hi, col1=sq_lo, col2=ones (row-mask counts).
// ---------------------------------------------------------------------------
__global__ __launch_bounds__(256) void edge_kernel(
    const float* __restrict__ adj,
    const float* __restrict__ s, const float* __restrict__ t,
    const float* __restrict__ sqArr,
    const unsigned short* __restrict__ dHiT, const unsigned short* __restrict__ dLoT,
    const unsigned short* __restrict__ sqHi, const unsigned short* __restrict__ sqLo,
    const unsigned short* __restrict__ onesBuf, const unsigned short* __restrict__ zeroBuf,
    float* __restrict__ acc)
{
    const int tid  = threadIdx.x;
    const int wv   = tid >> 5;
    const int lane = tid & 31;
    const int r    = lane & 15;     // A row within tile; also N index of B/C
    const int half = lane >> 4;     // 0: lanes 0-15, 1: lanes 16-31
    const int rowBase = (blockIdx.x * 8 + wv) * 16;
    const int kStart  = blockIdx.y * (N_NODES / K_SPLIT);
    const int kEnd    = kStart + (N_NODES / K_SPLIT);
    const int aOff = half * 16;     // A lane K-base: {0..15} vs {16..31}
    const int bOff = half * 16;     // B lane K-base: {0..15} vs {16..31}

    // per-lane column pointer for the extra tile (uniform control flow in loop)
    const unsigned short* xPtr = (r == 0) ? sqHi
                               : (r == 1) ? sqLo
                               : (r == 2) ? onesBuf : zeroBuf;

    v8f C[8];
    v8f Cx;
    #pragma unroll
    for (int nt = 0; nt < 8; ++nt)
        #pragma unroll
        for (int q = 0; q < 8; ++q) C[nt][q] = 0.0f;
    #pragma unroll
    for (int q = 0; q < 8; ++q) Cx[q] = 0.0f;

    v2f adjS;
    adjS.x = 0.0f; adjS.y = 0.0f;
    const float* aRow = adj + (size_t)(rowBase + r) * N_NODES;

    for (int kc = kStart; kc < kEnd; kc += 32) {
        // ---- A operand: 16x32 mask tile (contiguous k = 16*half + s layout) ----
        const float* ap = aRow + kc + aOff;
        __builtin_prefetch(ap + 256, 0, 0);          // next 8 chunks of this row
        float4 a0 = *(const float4*)(ap);
        float4 a1 = *(const float4*)(ap + 4);
        float4 a2 = *(const float4*)(ap + 8);
        float4 a3 = *(const float4*)(ap + 12);
        float av[16] = {a0.x,a0.y,a0.z,a0.w, a1.x,a1.y,a1.z,a1.w,
                        a2.x,a2.y,a2.z,a2.w, a3.x,a3.y,a3.z,a3.w};
        V16U mu;
        #pragma unroll
        for (int j = 0; j < 8; ++j) {
            float x0 = av[2 * j], x1 = av[2 * j + 1];
            v2f p2; p2.x = x0; p2.y = x1;
            adjS += p2;
            unsigned p = (x0 > 0.0f ? 0x00003F80u : 0u) |
                         (x1 > 0.0f ? 0x3F800000u : 0u);   // packed bf16 {1.0,1.0}
            mu.u[j] = p;
        }
        v16bf am = mu.v;

        // ---- 8 feature tiles, hi + lo bf16 GEMMs (f32-accurate) ----
        #pragma unroll
        for (int nt = 0; nt < 8; ++nt) {
            size_t boff = (size_t)(nt * 16 + r) * N_NODES + kc + bOff;
            C[nt] = wmma_bf16(am, load_v16(dHiT + boff), C[nt]);
            C[nt] = wmma_bf16(am, load_v16(dLoT + boff), C[nt]);
        }

        // ---- extra tile: col0=sq_hi, col1=sq_lo, col2=ones, rest zero ----
        Cx = wmma_bf16(am, load_v16(xPtr + kc + bOff), Cx);
    }

    // ---- dot(C, d) with exact f32 d; C layout: VGPR v -> M = v + 8*half, N = r
    float dot = 0.0f;
    #pragma unroll
    for (int nt = 0; nt < 8; ++nt) {
        const int f = nt * 16 + r;
        #pragma unroll
        for (int v = 0; v < 8; ++v) {
            const int i = rowBase + v + half * 8;
            const size_t idx = (size_t)i * D_FEAT + f;
            float d = s[idx] - t[idx];
            dot += C[nt][v] * d;
        }
    }

    float term = 0.0f;
    if (r < 2) {               // cols 0,1: sum_i (M @ sq)_i
        #pragma unroll
        for (int v = 0; v < 8; ++v) term += Cx[v];
    } else if (r == 2) {       // col 2: row-mask counts -> sum_i sq_i * cnt_i
        #pragma unroll
        for (int v = 0; v < 8; ++v) term += Cx[v] * sqArr[rowBase + v + half * 8];
    }

    float partial = term - 2.0f * dot;
    float adjSum  = adjS.x + adjS.y;

    #pragma unroll
    for (int m = 16; m > 0; m >>= 1) {
        partial += __shfl_xor(partial, m, 32);
        adjSum  += __shfl_xor(adjSum,  m, 32);
    }
    if (lane == 0) {
        atomicAdd(&acc[1], partial);
        atomicAdd(&acc[0], adjSum);
    }
}

// ---------------------------------------------------------------------------
// Kernel 3: combine
// ---------------------------------------------------------------------------
__global__ void final_kernel(const float* __restrict__ acc, float* __restrict__ out) {
    float adjSum  = acc[0];
    float edgeNum = acc[1];
    float totSq   = acc[2];
    float node = totSq / (float)(N_NODES * D_FEAT);
    float edge = (edgeNum / (float)D_FEAT) / (adjSum + EPS_F);
    out[0] = 1.0f * (node + edge);   // WEIGHT = 1.0
}

extern "C" void kernel_launch(void* const* d_in, const int* in_sizes, int n_in,
                              void* d_out, int out_size, void* d_ws, size_t ws_size,
                              hipStream_t stream) {
    (void)in_sizes; (void)n_in; (void)out_size; (void)ws_size;
    const float* s   = (const float*)d_in[0];
    const float* t   = (const float*)d_in[1];
    const float* adj = (const float*)d_in[2];

    char* ws = (char*)d_ws;
    float* acc             = (float*)ws;                        // accumulators
    float* sqArr           = (float*)(ws + 256);                // 8192 f32   (32 KB)
    unsigned short* sqHi   = (unsigned short*)(ws + 33024);     // 8192 bf16  (16 KB)
    unsigned short* sqLo   = (unsigned short*)(ws + 49408);     // 8192 bf16
    unsigned short* zeroB  = (unsigned short*)(ws + 65792);     // 8192 bf16 zeros
    unsigned short* onesB  = (unsigned short*)(ws + 82176);     // 8192 bf16 ones
    unsigned short* dHiT   = (unsigned short*)(ws + 98560);     // 2 MB
    unsigned short* dLoT   = dHiT + (size_t)D_FEAT * N_NODES;   // 2 MB

    hipMemsetAsync(ws, 0, 256, stream);                 // accumulators
    hipMemsetAsync(ws + 65792, 0, 16384, stream);       // zero column buffer
    prep_kernel<<<N_NODES, D_FEAT, 0, stream>>>(s, t, sqArr, dHiT, dLoT,
                                                sqHi, sqLo, onesB, acc);
    edge_kernel<<<dim3(N_NODES / (16 * 8), K_SPLIT), 256, 0, stream>>>(
        adj, s, t, sqArr, dHiT, dLoT, sqHi, sqLo, onesB, zeroB, acc);
    final_kernel<<<1, 1, 0, stream>>>(acc, (float*)d_out);
}
